// Masked_MHA_LORA_57329223467029
// MI455X (gfx1250) — matrix operand order repaired and checked by reading, not compiled, not run on hardware
//
#include <hip/hip_runtime.h>

#define B_  2
#define S_  2048
#define E_  1024
#define H_  16
#define DK_ 64
#define R_  8

typedef __bf16 bf16_t;
typedef __attribute__((ext_vector_type(16))) __bf16 v16bf;
typedef __attribute__((ext_vector_type(8)))  __bf16 v8bf;
typedef __attribute__((ext_vector_type(8)))  float  v8f;
typedef __attribute__((ext_vector_type(4)))  unsigned int u32x4;
typedef __attribute__((ext_vector_type(8)))  int i32x8;
typedef __attribute__((ext_vector_type(4)))  int i32x4;

// Build a 16-element bf16 A/B operand from two 8-element (16B-aligned) chunks.
static __device__ __forceinline__ v16bf pack2(const bf16_t* p0, const bf16_t* p1) {
    v8bf a = *(const v8bf*)p0;
    v8bf b = *(const v8bf*)p1;
    v16bf r;
#pragma unroll
    for (int i = 0; i < 8; ++i) { r[i] = a[i]; r[i + 8] = b[i]; }
    return r;
}

static __device__ __forceinline__ v8f wmma_bf16(v16bf a, v16bf b, v8f c) {
    // D = A(16x32 bf16) * B(32x16 bf16) + C(16x16 f32)
    return __builtin_amdgcn_wmma_f32_16x16x32_bf16(false, a, false, b, (short)0, c,
                                                   false, false);
}

// ---------------------------------------------------------------------------
// Tensor Data Mover: 2-D bf16 tile (tile_d0 x tile_d1 elements, row stride
// stride_d0 elements) from global -> LDS.  D# per CDNA5 ISA ch.8.
// ---------------------------------------------------------------------------
static __device__ __forceinline__ void tdm_load_2d_bf16(
    unsigned lds_off, const void* gaddr,
    unsigned tile_d0, unsigned tile_d1,
    unsigned tensor_d0, unsigned tensor_d1, unsigned stride_d0) {
    unsigned long long ga = (unsigned long long)gaddr;
    u32x4 g0 = 0;
    g0[0] = 1u;                                                 // count=1
    g0[1] = lds_off;                                            // lds_addr
    g0[2] = (unsigned)(ga & 0xFFFFFFFFu);                       // global_addr lo
    g0[3] = (unsigned)((ga >> 32) & 0x01FFFFFFu) | (2u << 30);  // addr hi | type=2
    i32x8 g1 = 0;
    g1[0] = (int)(1u << 16);                                    // data_size = 2B
    g1[1] = (int)((tensor_d0 & 0xFFFFu) << 16);                 // tensor_dim0 lo16
    g1[2] = (int)(((tensor_d0 >> 16) & 0xFFFFu) |
                  ((tensor_d1 & 0xFFFFu) << 16));               // dim0 hi | dim1 lo
    g1[3] = (int)(((tensor_d1 >> 16) & 0xFFFFu) |
                  ((tile_d0 & 0xFFFFu) << 16));                 // dim1 hi | tile0
    g1[4] = (int)(tile_d1 & 0xFFFFu);                           // tile1, tile2=0
    g1[5] = (int)stride_d0;                                     // dim0_stride lo32
    g1[6] = 0;
    g1[7] = 0;
    i32x4 z4 = 0;
#if __clang_major__ >= 23
    i32x8 z8 = 0;
    __builtin_amdgcn_tensor_load_to_lds(g0, g1, z4, z4, z8, 0);
#else
    __builtin_amdgcn_tensor_load_to_lds(g0, g1, z4, z4, 0);
#endif
}

// ---------------------------------------------------------------------------
// Kernel 1: RoPE cos/sin tables [S, DK]
// ---------------------------------------------------------------------------
__global__ __launch_bounds__(256) void rope_prep_kernel(float* __restrict__ cosT,
                                                        float* __restrict__ sinT) {
    int idx = blockIdx.x * 256 + threadIdx.x;           // S_*(DK_/2) threads
    int i = idx & 31;
    int s = idx >> 5;
    float inv = __powf(10000.0f, -(2.0f * (float)i) / (float)DK_);
    float ang = (float)s * inv;
    float c = __cosf(ang), sn = __sinf(ang);
    cosT[s * DK_ + i] = c;  cosT[s * DK_ + i + 32] = c;
    sinT[s * DK_ + i] = sn; sinT[s * DK_ + i + 32] = sn;
}

// ---------------------------------------------------------------------------
// Kernel 2: LoRA low-rank left factor  t[p][m][r] = x[m] . A_p[r]   (R = 8)
// ---------------------------------------------------------------------------
__global__ __launch_bounds__(256) void lora_t_kernel(const float* __restrict__ x,
                                                     const float* __restrict__ qA,
                                                     const float* __restrict__ kA,
                                                     const float* __restrict__ vA,
                                                     float* __restrict__ tBuf) {
    int idx = blockIdx.x * 256 + threadIdx.x;           // 3 * 4096 * 8 threads
    int r = idx & 7;
    int m = (idx >> 3) & 4095;
    int pj = idx >> 15;
    const float* A  = pj == 0 ? qA : (pj == 1 ? kA : vA);
    const float* xr = x + (size_t)m * E_;
    const float* ar = A + (size_t)r * E_;
    float acc = 0.f;
    for (int k = 0; k < E_; k += 4) {
        float4 xv = *(const float4*)(xr + k);
        float4 av = *(const float4*)(ar + k);
        acc += xv.x * av.x + xv.y * av.y + xv.z * av.z + xv.w * av.w;
    }
    tBuf[idx] = acc;
}

// ---------------------------------------------------------------------------
// Kernel 3: fused QKV projection GEMM + bias + LoRA + RoPE -> bf16 [B,H,S,DK]
// block = 256 thr (8 waves), tile 128(M) x 64(N), K-steps of 32 (bf16 WMMA)
// grid = (M/128, E/64 == H, 3)
// ---------------------------------------------------------------------------
__global__ __launch_bounds__(256) void qkv_gemm_rope_kernel(
    const float* __restrict__ x,
    const float* __restrict__ qW, const float* __restrict__ qb,
    const float* __restrict__ kW, const float* __restrict__ kb,
    const float* __restrict__ vW, const float* __restrict__ vb,
    const float* __restrict__ qB, const float* __restrict__ kB,
    const float* __restrict__ vB,
    const float* __restrict__ tBuf,
    const float* __restrict__ ropeCos, const float* __restrict__ ropeSin,
    bf16_t* __restrict__ Qr, bf16_t* __restrict__ Kr, bf16_t* __restrict__ Vr) {

    __shared__ bf16_t Alds[128][32];
    __shared__ bf16_t Blds[64][32];
    __shared__ float  Ctile[128][64];

    const int proj = blockIdx.z;
    const float* W    = proj == 0 ? qW : (proj == 1 ? kW : vW);
    const float* bias = proj == 0 ? qb : (proj == 1 ? kb : vb);
    const float* Bm   = proj == 0 ? qB : (proj == 1 ? kB : vB);
    bf16_t* dst       = proj == 0 ? Qr : (proj == 1 ? Kr : Vr);

    const int t = threadIdx.x;
    const int lane = t & 31, w = t >> 5;
    const int Mbase = blockIdx.x * 128;
    const int Nbase = blockIdx.y * 64;      // one full head per N-block
    const int wm = (w >> 1) * 32;           // wave M offset (0/32/64/96)
    const int wn = (w & 1) * 32;            // wave N offset (0/32)
    const int kA_ = (lane & 16) ? 8 : 0;    // A-layout K half
    const int kB_ = (lane & 16) ? 8 : 0;    // B-layout K half

    v8f acc[2][2] = {};

    for (int k0 = 0; k0 < E_; k0 += 32) {
        __syncthreads();
        {   // stage A: 128x32 f32 -> bf16 LDS (row t>>1, col half t&1)
            int row = t >> 1, col = (t & 1) * 16;
            const float* src = x + (size_t)(Mbase + row) * E_ + k0 + col;
            if (k0 + 32 < E_) __builtin_prefetch(src + 32, 0, 3);  // global_prefetch
            float4 f0 = *(const float4*)(src);
            float4 f1 = *(const float4*)(src + 4);
            float4 f2 = *(const float4*)(src + 8);
            float4 f3 = *(const float4*)(src + 12);
            v8bf p0, p1;
            p0[0]=(bf16_t)f0.x; p0[1]=(bf16_t)f0.y; p0[2]=(bf16_t)f0.z; p0[3]=(bf16_t)f0.w;
            p0[4]=(bf16_t)f1.x; p0[5]=(bf16_t)f1.y; p0[6]=(bf16_t)f1.z; p0[7]=(bf16_t)f1.w;
            p1[0]=(bf16_t)f2.x; p1[1]=(bf16_t)f2.y; p1[2]=(bf16_t)f2.z; p1[3]=(bf16_t)f2.w;
            p1[4]=(bf16_t)f3.x; p1[5]=(bf16_t)f3.y; p1[6]=(bf16_t)f3.z; p1[7]=(bf16_t)f3.w;
            *(v8bf*)&Alds[row][col]     = p0;
            *(v8bf*)&Alds[row][col + 8] = p1;
        }
        if (t < 128) {  // stage B: 64x32 of W (row-major [N][K]) -> bf16 LDS
            int row = t >> 1, col = (t & 1) * 16;
            const float* src = W + (size_t)(Nbase + row) * E_ + k0 + col;
            float4 f0 = *(const float4*)(src);
            float4 f1 = *(const float4*)(src + 4);
            float4 f2 = *(const float4*)(src + 8);
            float4 f3 = *(const float4*)(src + 12);
            v8bf p0, p1;
            p0[0]=(bf16_t)f0.x; p0[1]=(bf16_t)f0.y; p0[2]=(bf16_t)f0.z; p0[3]=(bf16_t)f0.w;
            p0[4]=(bf16_t)f1.x; p0[5]=(bf16_t)f1.y; p0[6]=(bf16_t)f1.z; p0[7]=(bf16_t)f1.w;
            p1[0]=(bf16_t)f2.x; p1[1]=(bf16_t)f2.y; p1[2]=(bf16_t)f2.z; p1[3]=(bf16_t)f2.w;
            p1[4]=(bf16_t)f3.x; p1[5]=(bf16_t)f3.y; p1[6]=(bf16_t)f3.z; p1[7]=(bf16_t)f3.w;
            *(v8bf*)&Blds[row][col]     = p0;
            *(v8bf*)&Blds[row][col + 8] = p1;
        }
        __syncthreads();
#pragma unroll
        for (int mt = 0; mt < 2; ++mt) {
            const bf16_t* ap = &Alds[wm + mt * 16 + (lane & 15)][kA_];
            v16bf a = pack2(ap, ap + 16);
#pragma unroll
            for (int nt = 0; nt < 2; ++nt) {
                const bf16_t* bp = &Blds[wn + nt * 16 + (lane & 15)][kB_];
                v16bf b = pack2(bp, bp + 16);
                acc[mt][nt] = wmma_bf16(a, b, acc[mt][nt]);
            }
        }
    }

    // epilogue: bias + scaled LoRA, stash f32 tile in LDS for RoPE pairing
    const float* tRow = tBuf + (size_t)proj * 4096 * R_;
#pragma unroll
    for (int mt = 0; mt < 2; ++mt)
#pragma unroll
        for (int nt = 0; nt < 2; ++nt)
#pragma unroll
            for (int r = 0; r < 8; ++r) {
                int Ml = wm + mt * 16 + r + ((lane & 16) ? 8 : 0);
                int Nl = wn + nt * 16 + (lane & 15);
                int Mg = Mbase + Ml, Ng = Nbase + Nl;
                const float* tp = tRow + (size_t)Mg * R_;
                const float* bp = Bm + (size_t)Ng * R_;
                float lacc = 0.f;
#pragma unroll
                for (int rr = 0; rr < R_; ++rr) lacc += tp[rr] * bp[rr];
                Ctile[Ml][Nl] = acc[mt][nt][r] + bias[Ng] + 2.0f * lacc; // SCALING=2
            }
    __syncthreads();

    // RoPE (Q,K) / pass-through (V), convert to bf16 in [B,H,S,DK]
    const int h = blockIdx.y;
    for (int idx = t; idx < 128 * 64; idx += 256) {
        int ml = idx >> 6, d = idx & 63;
        int Mg = Mbase + ml;
        int bb = Mg >> 11;         // / S_
        int s  = Mg & (S_ - 1);
        float v = Ctile[ml][d];
        if (proj < 2) {
            int pr = (d < 32) ? d + 32 : d - 32;
            float vp = Ctile[ml][pr];
            float c = ropeCos[s * DK_ + d], sn = ropeSin[s * DK_ + d];
            v = v * c + ((d < 32) ? -vp : vp) * sn;
        }
        dst[(((size_t)bb * H_ + h) * S_ + s) * DK_ + d] = (bf16_t)v;
    }
}

// ---------------------------------------------------------------------------
// Kernel 4: causal flash attention, one (b, h, 128 q-rows) per workgroup.
// K tile staged by the Tensor Data Mover; V^T staged manually (transpose).
// ---------------------------------------------------------------------------
__global__ __launch_bounds__(256) void attn_kernel(const bf16_t* __restrict__ Qr,
                                                   const bf16_t* __restrict__ Kr,
                                                   const bf16_t* __restrict__ Vr,
                                                   const int* __restrict__ amask,
                                                   bf16_t* __restrict__ Obf) {
    __shared__ bf16_t Klds[64][64];        // [key][dk]  (TDM destination)
    __shared__ bf16_t Vt[64][64];          // [dk][key]  (transposed)
    __shared__ bf16_t Plds[8][16][64];     // per-wave P tile

    const int t = threadIdx.x, lane = t & 31, w = t >> 5;
    const int qblk = blockIdx.x, h = blockIdx.y, b = blockIdx.z;
    const int qbase = qblk * 128;
    const size_t headOff = ((size_t)b * H_ + h) * S_ * DK_;
    const int* mk = amask + (size_t)b * S_;
    const unsigned kldsOff = (unsigned)(unsigned long long)(&Klds[0][0]);

    const int kA_ = (lane & 16) ? 8 : 0;
    const int kB_ = (lane & 16) ? 8 : 0;

    // Q stripe for this wave: A-layout operands, loaded once
    const int qrow = qbase + w * 16 + (lane & 15);
    const bf16_t* qp = Qr + headOff + (size_t)qrow * DK_;
    v16bf aq0 = pack2(qp + kA_, qp + kA_ + 16);
    v16bf aq1 = pack2(qp + 32 + kA_, qp + 32 + kA_ + 16);

    v8f o[4] = {};
    float mrow[8], lrow[8];
#pragma unroll
    for (int r = 0; r < 8; ++r) { mrow[r] = -3e38f; lrow[r] = 0.f; }

    const int nj = 2 * qblk + 2;           // causal: key blocks of 64
    for (int j = 0; j < nj; ++j) {
        __syncthreads();                   // all prior LDS readers done
        if (w == 0) {                      // TDM: 64x64 bf16 K tile -> LDS
            tdm_load_2d_bf16(kldsOff, Kr + headOff + (size_t)j * 64 * DK_,
                             /*tile*/ DK_, 64, /*tensor*/ DK_, S_, /*stride*/ DK_);
        }
        {   // stage V transposed (manual: TDM cannot transpose)
            int key = t >> 2;
            int c0 = (t & 3) * 16;
            const bf16_t* vs = Vr + headOff + (size_t)(j * 64 + key) * DK_ + c0;
            v8bf v0 = *(const v8bf*)vs;
            v8bf v1 = *(const v8bf*)(vs + 8);
#pragma unroll
            for (int e = 0; e < 8; ++e) {
                Vt[c0 + e][key]     = v0[e];
                Vt[c0 + 8 + e][key] = v1[e];
            }
        }
        if (w == 0) __builtin_amdgcn_s_wait_tensorcnt(0);
        __syncthreads();

        // scores S = Q K^T (16 x 64), scale + causal/pad mask
        v8f sc[4];
#pragma unroll
        for (int nt = 0; nt < 4; ++nt) {
            const bf16_t* b0 = &Klds[nt * 16 + (lane & 15)][kB_];
            const bf16_t* b1 = &Klds[nt * 16 + (lane & 15)][32 + kB_];
            v8f z = 0;
            v8f c0 = wmma_bf16(aq0, pack2(b0, b0 + 16), z);
            sc[nt]  = wmma_bf16(aq1, pack2(b1, b1 + 16), c0);
        }
        const int qg0 = qbase + w * 16 + ((lane & 16) ? 8 : 0);
#pragma unroll
        for (int nt = 0; nt < 4; ++nt) {
            int kg = j * 64 + nt * 16 + (lane & 15);
            bool pad_ok = (mk[kg] != 0);
#pragma unroll
            for (int r = 0; r < 8; ++r) {
                bool ok = (kg <= qg0 + r) && pad_ok;
                sc[nt][r] = ok ? sc[nt][r] * 0.125f : -1e30f;
            }
        }

        // online softmax: row max / sum via shfl within 16-lane halves
        v8f vm = sc[0];
#pragma unroll
        for (int nt = 1; nt < 4; ++nt)
#pragma unroll
            for (int r = 0; r < 8; ++r) vm[r] = fmaxf(vm[r], sc[nt][r]);
#pragma unroll
        for (int msk = 1; msk < 16; msk <<= 1)
#pragma unroll
            for (int r = 0; r < 8; ++r)
                vm[r] = fmaxf(vm[r], __shfl_xor(vm[r], msk, 32));

        float scl[8];
#pragma unroll
        for (int r = 0; r < 8; ++r) {
            float nm = fmaxf(mrow[r], vm[r]);
            scl[r] = __expf(mrow[r] - nm);
            mrow[r] = nm;
        }
        v8f vs = 0;
#pragma unroll
        for (int nt = 0; nt < 4; ++nt)
#pragma unroll
            for (int r = 0; r < 8; ++r) {
                float p = __expf(sc[nt][r] - mrow[r]);
                sc[nt][r] = p;
                vs[r] += p;
            }
#pragma unroll
        for (int msk = 1; msk < 16; msk <<= 1)
#pragma unroll
            for (int r = 0; r < 8; ++r) vs[r] += __shfl_xor(vs[r], msk, 32);
#pragma unroll
        for (int r = 0; r < 8; ++r) lrow[r] = lrow[r] * scl[r] + vs[r];
#pragma unroll
        for (int nt = 0; nt < 4; ++nt)
#pragma unroll
            for (int r = 0; r < 8; ++r) o[nt][r] *= scl[r];

        // P -> LDS (bf16) to re-layout C-matrix -> A-matrix
        const int pr0 = (lane & 16) ? 8 : 0;
#pragma unroll
        for (int nt = 0; nt < 4; ++nt)
#pragma unroll
            for (int r = 0; r < 8; ++r)
                Plds[w][pr0 + r][nt * 16 + (lane & 15)] = (bf16_t)sc[nt][r];
        __syncthreads();

        // O += P @ V
        const bf16_t* pb = &Plds[w][lane & 15][0];
        v16bf ap0 = pack2(pb + kA_, pb + kA_ + 16);
        v16bf ap1 = pack2(pb + 32 + kA_, pb + 32 + kA_ + 16);
#pragma unroll
        for (int nt = 0; nt < 4; ++nt) {
            const bf16_t* vb0 = &Vt[nt * 16 + (lane & 15)][kB_];
            const bf16_t* vb1 = &Vt[nt * 16 + (lane & 15)][32 + kB_];
            o[nt] = wmma_bf16(ap0, pack2(vb0, vb0 + 16), o[nt]);
            o[nt] = wmma_bf16(ap1, pack2(vb1, vb1 + 16), o[nt]);
        }
    }

    // normalize and write bf16 [B, S, E] (head-interleaved for out-proj GEMM)
#pragma unroll
    for (int nt = 0; nt < 4; ++nt)
#pragma unroll
        for (int r = 0; r < 8; ++r) {
            int qg = qbase + w * 16 + ((lane & 16) ? 8 : 0) + r;
            size_t oi = ((size_t)b * S_ + qg) * E_ + h * DK_ + nt * 16 + (lane & 15);
            Obf[oi] = (bf16_t)(o[nt][r] / lrow[r]);
        }
}

// ---------------------------------------------------------------------------
// Kernel 5: output projection  out = Obf @ oW^T + ob  (f32 result)
// A tile (bf16, raw copy) staged by the Tensor Data Mover.
// ---------------------------------------------------------------------------
__global__ __launch_bounds__(256) void out_gemm_kernel(const bf16_t* __restrict__ Obf,
                                                       const float* __restrict__ oW,
                                                       const float* __restrict__ ob,
                                                       float* __restrict__ out) {
    __shared__ bf16_t Alds[128][32];
    __shared__ bf16_t Blds[64][32];

    const int t = threadIdx.x, lane = t & 31, w = t >> 5;
    const int Mbase = blockIdx.x * 128;
    const int Nbase = blockIdx.y * 64;
    const int wm = (w >> 1) * 32;
    const int wn = (w & 1) * 32;
    const int kA_ = (lane & 16) ? 8 : 0;
    const int kB_ = (lane & 16) ? 8 : 0;
    const unsigned aldsOff = (unsigned)(unsigned long long)(&Alds[0][0]);

    v8f acc[2][2] = {};

    for (int k0 = 0; k0 < E_; k0 += 32) {
        __syncthreads();
        if (w == 0) {   // TDM: 128 rows x 32 bf16 from Obf[Mbase..][k0..]
            tdm_load_2d_bf16(aldsOff, Obf + (size_t)Mbase * E_ + k0,
                             /*tile*/ 32, 128, /*tensor*/ E_, B_ * S_, /*stride*/ E_);
        }
        if (t < 128) {  // B: oW f32 -> bf16 (needs conversion: manual)
            int row = t >> 1, col = (t & 1) * 16;
            const float* src = oW + (size_t)(Nbase + row) * E_ + k0 + col;
            if (k0 + 32 < E_) __builtin_prefetch(src + 32, 0, 3);
            float4 f0 = *(const float4*)(src);
            float4 f1 = *(const float4*)(src + 4);
            float4 f2 = *(const float4*)(src + 8);
            float4 f3 = *(const float4*)(src + 12);
            v8bf p0, p1;
            p0[0]=(bf16_t)f0.x; p0[1]=(bf16_t)f0.y; p0[2]=(bf16_t)f0.z; p0[3]=(bf16_t)f0.w;
            p0[4]=(bf16_t)f1.x; p0[5]=(bf16_t)f1.y; p0[6]=(bf16_t)f1.z; p0[7]=(bf16_t)f1.w;
            p1[0]=(bf16_t)f2.x; p1[1]=(bf16_t)f2.y; p1[2]=(bf16_t)f2.z; p1[3]=(bf16_t)f2.w;
            p1[4]=(bf16_t)f3.x; p1[5]=(bf16_t)f3.y; p1[6]=(bf16_t)f3.z; p1[7]=(bf16_t)f3.w;
            *(v8bf*)&Blds[row][col]     = p0;
            *(v8bf*)&Blds[row][col + 8] = p1;
        }
        if (w == 0) __builtin_amdgcn_s_wait_tensorcnt(0);
        __syncthreads();
#pragma unroll
        for (int mt = 0; mt < 2; ++mt) {
            const bf16_t* ap = &Alds[wm + mt * 16 + (lane & 15)][kA_];
            v16bf a = pack2(ap, ap + 16);
#pragma unroll
            for (int nt = 0; nt < 2; ++nt) {
                const bf16_t* bp = &Blds[wn + nt * 16 + (lane & 15)][kB_];
                v16bf b = pack2(bp, bp + 16);
                acc[mt][nt] = wmma_bf16(a, b, acc[mt][nt]);
            }
        }
    }

#pragma unroll
    for (int mt = 0; mt < 2; ++mt)
#pragma unroll
        for (int nt = 0; nt < 2; ++nt)
#pragma unroll
            for (int r = 0; r < 8; ++r) {
                int Mg = Mbase + wm + mt * 16 + r + ((lane & 16) ? 8 : 0);
                int Ng = Nbase + wn + nt * 16 + (lane & 15);
                out[(size_t)Mg * E_ + Ng] = acc[mt][nt][r] + ob[Ng];
            }
}

// ---------------------------------------------------------------------------
extern "C" void kernel_launch(void* const* d_in, const int* in_sizes, int n_in,
                              void* d_out, int out_size, void* d_ws, size_t ws_size,
                              hipStream_t stream) {
    (void)in_sizes; (void)n_in; (void)out_size; (void)ws_size;
    const float* x  = (const float*)d_in[0];
    const int* am   = (const int*)d_in[1];
    const float* qW = (const float*)d_in[2];
    const float* qb = (const float*)d_in[3];
    const float* kW = (const float*)d_in[4];
    const float* kb = (const float*)d_in[5];
    const float* vW = (const float*)d_in[6];
    const float* vb = (const float*)d_in[7];
    const float* qA = (const float*)d_in[8];
    const float* qB = (const float*)d_in[9];
    const float* kA = (const float*)d_in[10];
    const float* kB = (const float*)d_in[11];
    const float* vA = (const float*)d_in[12];
    const float* vB = (const float*)d_in[13];
    const float* oW = (const float*)d_in[14];
    const float* ob = (const float*)d_in[15];
    float* out = (float*)d_out;

    char* p = (char*)d_ws;
    float* ropeCos = (float*)p; p += (size_t)S_ * DK_ * 4;
    float* ropeSin = (float*)p; p += (size_t)S_ * DK_ * 4;
    float* tBuf    = (float*)p; p += (size_t)3 * B_ * S_ * R_ * 4;
    bf16_t* Qr  = (bf16_t*)p;   p += (size_t)B_ * H_ * S_ * DK_ * 2;
    bf16_t* Kr  = (bf16_t*)p;   p += (size_t)B_ * H_ * S_ * DK_ * 2;
    bf16_t* Vr  = (bf16_t*)p;   p += (size_t)B_ * H_ * S_ * DK_ * 2;
    bf16_t* Obf = (bf16_t*)p;   p += (size_t)B_ * S_ * E_ * 2;

    rope_prep_kernel<<<dim3((S_ * (DK_ / 2)) / 256), dim3(256), 0, stream>>>(ropeCos, ropeSin);
    lora_t_kernel<<<dim3((3 * B_ * S_ * R_) / 256), dim3(256), 0, stream>>>(x, qA, kA, vA, tBuf);
    qkv_gemm_rope_kernel<<<dim3((B_ * S_) / 128, E_ / 64, 3), dim3(256), 0, stream>>>(
        x, qW, qb, kW, kb, vW, vb, qB, kB, vB, tBuf, ropeCos, ropeSin, Qr, Kr, Vr);
    attn_kernel<<<dim3(S_ / 128, H_, B_), dim3(256), 0, stream>>>(Qr, Kr, Vr, am, Obf);
    out_gemm_kernel<<<dim3((B_ * S_) / 128, E_ / 64), dim3(256), 0, stream>>>(Obf, oW, ob, out);
}
